// Net_9311489097854
// MI455X (gfx1250) — compile-verified, matrix operand order checked by reading, not run on hardware
//
#include <hip/hip_runtime.h>
#include <hip/hip_bf16.h>
#include <math.h>

// ---------------- problem constants (match reference) ----------------
#define NN 100000      // nodes
#define FF 128         // input features
#define HH 256         // hidden
#define CC 64          // classes
#define EE 1600000     // edges
#define KPROP 10
#define ALPHA_ 0.1f
#define BETA_  0.5f
#define EPS_   1e-5f

// ---------------- types for WMMA ----------------
typedef __bf16 bf16_t;
typedef __attribute__((ext_vector_type(16))) bf16_t v16bf;
typedef __attribute__((ext_vector_type(8)))  bf16_t bfx8;
typedef __attribute__((ext_vector_type(4)))  bf16_t bfx4;
typedef __attribute__((ext_vector_type(8)))  float  v8f;

// =====================================================================
// 1) fp32 -> bf16 conversion (4 elems / thread, vectorized)
// =====================================================================
__global__ __launch_bounds__(256)
void k_f32_to_bf16_4(const float* __restrict__ in, bf16_t* __restrict__ out, int n4) {
    int i = blockIdx.x * blockDim.x + threadIdx.x;
    if (i >= n4) return;
    float4 v = ((const float4*)in)[i];
    bfx4 o;
    o[0] = (bf16_t)v.x; o[1] = (bf16_t)v.y; o[2] = (bf16_t)v.z; o[3] = (bf16_t)v.w;
    ((bfx4*)out)[i] = o;
}

// =====================================================================
// 2) pack fp32 row-major weight [K,Nout] into bf16 WMMA B-fragments.
// Fragment f = kc*(Nout/16)+nt holds 32 lanes x 16 bf16 contiguous.
// Element i of lane l maps to K = kc*32 + ((i>>3)<<4) + (l>=16?8:0) + (i&7),
// col = nt*16 + (l&15)  (mirror of the 16-bit A layout, lane = column).
// =====================================================================
__global__ __launch_bounds__(256)
void k_pack_w(const float* __restrict__ W, bf16_t* __restrict__ out, int K, int Nout) {
    int p = blockIdx.x * blockDim.x + threadIdx.x;
    if (p >= K * Nout) return;
    int f    = p >> 9;
    int r    = p & 511;
    int lane = r >> 4;
    int i    = r & 15;
    int ntn  = Nout >> 4;
    int kc   = f / ntn;
    int nt   = f - kc * ntn;
    int half = lane >> 4, idx = lane & 15;
    int kk   = kc * 32 + ((i >> 3) << 4) + (half << 3) + (i & 7);
    int col  = nt * 16 + idx;
    out[p] = (bf16_t)W[(size_t)kk * Nout + col];
}

// =====================================================================
// 3) WMMA GEMM: C[Nrows,Nout] = A(bf16,row-major)[Nrows,K] x W + bias
// Workgroup = 8 waves = 128x64 output block. The packed-B strip for the
// block's 64 columns (kch fragments x 4 tiles x 1KB) is staged into LDS
// once with async global->LDS copies (ASYNCcnt), then all 8 waves feed
// their WMMAs from LDS while A streams from global.
// =====================================================================
__global__ __launch_bounds__(256)
void k_gemm_wmma(const bf16_t* __restrict__ A, const bf16_t* __restrict__ Bp,
                 const float* __restrict__ bias, float* __restrict__ C,
                 int Nrows, int K, int Nout) {
    __shared__ bf16_t bsm[16384];            // up to 32 KB (kch<=8): kch*4 frags * 512 bf16

    const int ngrp  = Nout >> 6;             // groups of 4 n-tiles (64 cols)
    const int mtils = Nrows >> 4;
    const int mblk  = (mtils + 7) >> 3;      // 8 m-tiles per workgroup
    const int mb    = blockIdx.x / ngrp;
    const int ng    = blockIdx.x - mb * ngrp;
    const int ntn   = Nout >> 4;
    const int kch   = K >> 5;
    const int tid   = threadIdx.x;
    (void)mblk;

    // ---- async stage of B strip: kch chunks of 4KB (4 fragments each) ----
    {
        const int totalVec = kch << 8;       // 16B units: 256 per kc-chunk
        for (int v = tid; v < totalVec; v += 256) {
            int kc = v >> 8;
            int r  = v & 255;
            const bf16_t* gp = Bp + (((size_t)kc * ntn + ng * 4) << 9) + r * 8;
            unsigned long long ga = (unsigned long long)gp;
            unsigned loff = (unsigned)(unsigned long long)(bsm + (kc * 2048 + r * 8));
            asm volatile("global_load_async_to_lds_b128 %0, %1, off"
                         :: "v"(loff), "v"(ga) : "memory");
        }
        asm volatile("s_wait_asynccnt 0x0" ::: "memory");
        __syncthreads();
    }

    const int wv   = tid >> 5;
    const int lane = tid & 31;
    const int mt   = mb * 8 + wv;            // this wave's m-tile
    if (mt < mtils) {                        // wave-uniform guard (EXEC all-1 inside)
        const int half = lane >> 4, idx = lane & 15;

        v8f acc0 = {}, acc1 = {}, acc2 = {}, acc3 = {};
        const bf16_t* arow = A + (size_t)(mt * 16 + idx) * K + half * 8;

        for (int kc = 0; kc < kch; ++kc) {
            const bf16_t* ap = arow + kc * 32;
            bfx8 alo = *(const bfx8*)(ap);
            bfx8 ahi = *(const bfx8*)(ap + 16);
            v16bf a  = __builtin_shufflevector(alo, ahi,
                         0,1,2,3,4,5,6,7,8,9,10,11,12,13,14,15);

            const bf16_t* bp = bsm + kc * 2048 + lane * 16;   // LDS

            bfx8 b0l = *(const bfx8*)(bp);
            bfx8 b0h = *(const bfx8*)(bp + 8);
            v16bf b0 = __builtin_shufflevector(b0l, b0h, 0,1,2,3,4,5,6,7,8,9,10,11,12,13,14,15);
            acc0 = __builtin_amdgcn_wmma_f32_16x16x32_bf16(false, a, false, b0, (short)0, acc0, false, false);

            bfx8 b1l = *(const bfx8*)(bp + 512);
            bfx8 b1h = *(const bfx8*)(bp + 512 + 8);
            v16bf b1 = __builtin_shufflevector(b1l, b1h, 0,1,2,3,4,5,6,7,8,9,10,11,12,13,14,15);
            acc1 = __builtin_amdgcn_wmma_f32_16x16x32_bf16(false, a, false, b1, (short)0, acc1, false, false);

            bfx8 b2l = *(const bfx8*)(bp + 1024);
            bfx8 b2h = *(const bfx8*)(bp + 1024 + 8);
            v16bf b2 = __builtin_shufflevector(b2l, b2h, 0,1,2,3,4,5,6,7,8,9,10,11,12,13,14,15);
            acc2 = __builtin_amdgcn_wmma_f32_16x16x32_bf16(false, a, false, b2, (short)0, acc2, false, false);

            bfx8 b3l = *(const bfx8*)(bp + 1536);
            bfx8 b3h = *(const bfx8*)(bp + 1536 + 8);
            v16bf b3 = __builtin_shufflevector(b3l, b3h, 0,1,2,3,4,5,6,7,8,9,10,11,12,13,14,15);
            acc3 = __builtin_amdgcn_wmma_f32_16x16x32_bf16(false, a, false, b3, (short)0, acc3, false, false);
        }

        const int cb = ng * 64 + idx;
        const float bz0 = bias[cb], bz1 = bias[cb + 16], bz2 = bias[cb + 32], bz3 = bias[cb + 48];
#pragma unroll
        for (int j = 0; j < 8; ++j) {
            int row = mt * 16 + half * 8 + j;   // C/D layout: M = j + 8*half, N = idx
            float* cp = C + (size_t)row * Nout + cb;
            cp[0]  = acc0[j] + bz0;
            cp[16] = acc1[j] + bz1;
            cp[32] = acc2[j] + bz2;
            cp[48] = acc3[j] + bz3;
        }
    }
}

// =====================================================================
// 4) BatchNorm column stats over N rows (H=256 fixed): sums + sumsq
// =====================================================================
__global__ __launch_bounds__(256)
void k_zero_f32(float* __restrict__ p, int n) {
    int i = blockIdx.x * blockDim.x + threadIdx.x;
    if (i < n) p[i] = 0.0f;
}

__global__ __launch_bounds__(256)
void k_bn_stats(const float* __restrict__ h, float* __restrict__ stats, int Nrows) {
    int col = threadIdx.x;   // blockDim.x == 256 == HH
    float s = 0.0f, ss = 0.0f;
    for (int r = blockIdx.x; r < Nrows; r += gridDim.x) {
        float v = h[(size_t)r * HH + col];
        s += v; ss += v * v;
    }
    atomicAdd(&stats[col], s);
    atomicAdd(&stats[HH + col], ss);
}

__global__ __launch_bounds__(256)
void k_bn_relu_bf16(const float* __restrict__ h, const float* __restrict__ stats,
                    const float* __restrict__ g, const float* __restrict__ be,
                    bf16_t* __restrict__ out, int n) {
    int i = blockIdx.x * blockDim.x + threadIdx.x;
    if (i >= n) return;
    int c = i & (HH - 1);
    const float invN = 1.0f / (float)NN;
    float mu  = stats[c] * invN;
    float var = stats[HH + c] * invN - mu * mu;
    float v   = (h[i] - mu) * rsqrtf(var + EPS_) * g[c] + be[c];
    out[i] = (bf16_t)fmaxf(v, 0.0f);
}

// =====================================================================
// 5) warm-start blend:  z0 = any(z_sam!=0) ? 0.5*z_sam + 0.5*logits : logits
// =====================================================================
__global__ __launch_bounds__(256)
void k_zero_int(int* p) { if (threadIdx.x == 0 && blockIdx.x == 0) *p = 0; }

__global__ __launch_bounds__(256)
void k_any_nonzero(const float* __restrict__ p, int n, int* __restrict__ flag) {
    int i = blockIdx.x * blockDim.x + threadIdx.x;
    bool nz = (i < n) && (p[i] != 0.0f);
    unsigned long long m = __ballot(nz);
    if ((threadIdx.x & 31) == 0 && m) atomicOr(flag, 1);
}

__global__ __launch_bounds__(256)
void k_blend(const float* __restrict__ zs, const float* __restrict__ lg,
             const int* __restrict__ flag, float* __restrict__ z0, int n) {
    int i = blockIdx.x * blockDim.x + threadIdx.x;
    if (i >= n) return;
    float l = lg[i];
    z0[i] = (*flag) ? ((1.0f - BETA_) * zs[i] + BETA_ * l) : l;
}

// =====================================================================
// 6) propagation step:  zout = ALPHA*logits  then  zout[d] += 0.9*w*zin[s]
// =====================================================================
__global__ __launch_bounds__(256)
void k_prop_init(const float* __restrict__ lg, float* __restrict__ z, int n) {
    int i = blockIdx.x * blockDim.x + threadIdx.x;
    if (i < n) z[i] = ALPHA_ * lg[i];
}

__global__ __launch_bounds__(256)
void k_prop_scatter(const int* __restrict__ src, const int* __restrict__ dst,
                    const float* __restrict__ w, const float* __restrict__ zin,
                    float* __restrict__ acc) {
    int t = blockIdx.x * blockDim.x + threadIdx.x;   // EE*16 threads
    int e = t >> 4;
    if (e >= EE) return;
    int quad = t & 15;                                // 4 channels per thread
    int s = src[e];
    int d = dst[e];
    float sc = (1.0f - ALPHA_) * w[e];
    float4 zv = ((const float4*)(zin + (size_t)s * CC))[quad];
    float* ap = acc + (size_t)d * CC + quad * 4;
    atomicAdd(ap + 0, sc * zv.x);
    atomicAdd(ap + 1, sc * zv.y);
    atomicAdd(ap + 2, sc * zv.z);
    atomicAdd(ap + 3, sc * zv.w);
}

// =====================================================================
// 7) row log-softmax, C=64, one wave32 per row (2 cols / lane)
// =====================================================================
__global__ __launch_bounds__(256)
void k_log_softmax(const float* __restrict__ z, float* __restrict__ out, int Nrows) {
    int wave = threadIdx.x >> 5;
    int lane = threadIdx.x & 31;
    int row  = blockIdx.x * (blockDim.x >> 5) + wave;
    if (row >= Nrows) return;
    const float* zr = z + (size_t)row * CC;
    float v0 = zr[lane], v1 = zr[lane + 32];
    float m = fmaxf(v0, v1);
#pragma unroll
    for (int off = 16; off > 0; off >>= 1) m = fmaxf(m, __shfl_xor(m, off, 32));
    float s = __expf(v0 - m) + __expf(v1 - m);
#pragma unroll
    for (int off = 16; off > 0; off >>= 1) s += __shfl_xor(s, off, 32);
    float lse = m + __logf(s);
    float* orow = out + (size_t)row * CC;
    orow[lane]      = v0 - lse;
    orow[lane + 32] = v1 - lse;
}

// =====================================================================
// host-side orchestration
// =====================================================================
extern "C" void kernel_launch(void* const* d_in, const int* in_sizes, int n_in,
                              void* d_out, int out_size, void* d_ws, size_t ws_size,
                              hipStream_t stream) {
    (void)in_sizes; (void)n_in; (void)out_size; (void)ws_size;
    const float* x    = (const float*)d_in[0];
    const int*   src  = (const int*)  d_in[1];
    const int*   dstv = (const int*)  d_in[2];
    const float* w    = (const float*)d_in[3];
    const float* W1   = (const float*)d_in[4];
    const float* b1   = (const float*)d_in[5];
    const float* g1   = (const float*)d_in[6];
    const float* be1  = (const float*)d_in[7];
    const float* W2   = (const float*)d_in[8];
    const float* b2   = (const float*)d_in[9];
    const float* g2   = (const float*)d_in[10];
    const float* be2  = (const float*)d_in[11];
    const float* W3   = (const float*)d_in[12];
    const float* b3   = (const float*)d_in[13];
    const float* zsam = (const float*)d_in[14];
    float* out = (float*)d_out;

    // ---- workspace layout (with aliasing; ~179.5 MB) ----
    char* ws = (char*)d_ws;
    const size_t SZ_XBF  = (size_t)NN * FF * 2;   // 25.6 MB, bf16 x   (dead after GEMM1)
    const size_t SZ_HF32 = (size_t)NN * HH * 4;   // 102.4 MB, pre-BN f32 (dead after BN2)
    const size_t SZ_HBF  = (size_t)NN * HH * 2;   // 51.2 MB, bf16 activations
    bf16_t* xbf    = (bf16_t*)(ws);
    float*  logits = (float*) (ws);                         // alias xbf (used from GEMM3 on)
    float*  hf32   = (float*) (ws + SZ_XBF);
    float*  zA     = hf32;                                  // alias hf32 (used after BN2)
    float*  zB     = (float*) (ws + SZ_XBF + (size_t)NN * CC * 4);
    bf16_t* hbf    = (bf16_t*)(ws + SZ_XBF + SZ_HF32);
    bf16_t* W1p    = (bf16_t*)(ws + SZ_XBF + SZ_HF32 + SZ_HBF);
    bf16_t* W2p    = (bf16_t*)((char*)W1p + (size_t)FF * HH * 2);
    bf16_t* W3p    = (bf16_t*)((char*)W2p + (size_t)HH * HH * 2);
    float*  stats  = (float*) ((char*)W3p + (size_t)HH * CC * 2);
    int*    flag   = (int*)   ((char*)stats + 2 * HH * 4);

    dim3 B(256);
    const int mblk = ((NN / 16) + 7) / 8;    // 782 workgroups along M

    // 1) x -> bf16
    k_f32_to_bf16_4<<<(NN * FF / 4 + 255) / 256, B, 0, stream>>>(x, xbf, NN * FF / 4);

    // 2) pack weights into WMMA fragments
    k_pack_w<<<(FF * HH + 255) / 256, B, 0, stream>>>(W1, W1p, FF, HH);
    k_pack_w<<<(HH * HH + 255) / 256, B, 0, stream>>>(W2, W2p, HH, HH);
    k_pack_w<<<(HH * CC + 255) / 256, B, 0, stream>>>(W3, W3p, HH, CC);

    // 3) GEMM1: h1 = x @ W1 + b1
    k_gemm_wmma<<<mblk * (HH / 64), B, 0, stream>>>(xbf, W1p, b1, hf32, NN, FF, HH);
    // BN1 + ReLU -> bf16
    k_zero_f32<<<2, B, 0, stream>>>(stats, 2 * HH);
    k_bn_stats<<<1024, B, 0, stream>>>(hf32, stats, NN);
    k_bn_relu_bf16<<<(NN * HH + 255) / 256, B, 0, stream>>>(hf32, stats, g1, be1, hbf, NN * HH);

    // 4) GEMM2: h2 = h1 @ W2 + b2
    k_gemm_wmma<<<mblk * (HH / 64), B, 0, stream>>>(hbf, W2p, b2, hf32, NN, HH, HH);
    // BN2 + ReLU -> bf16 (overwrites hbf; reads only hf32)
    k_zero_f32<<<2, B, 0, stream>>>(stats, 2 * HH);
    k_bn_stats<<<1024, B, 0, stream>>>(hf32, stats, NN);
    k_bn_relu_bf16<<<(NN * HH + 255) / 256, B, 0, stream>>>(hf32, stats, g2, be2, hbf, NN * HH);

    // 5) GEMM3: logits = h2 @ W3 + b3   (logits aliases dead xbf region)
    k_gemm_wmma<<<mblk * (CC / 64), B, 0, stream>>>(hbf, W3p, b3, logits, NN, HH, CC);

    // 6) warm-start blend into zA  (zA/zB alias dead hf32 region)
    k_zero_int<<<1, 32, 0, stream>>>(flag);
    k_any_nonzero<<<(NN * CC + 255) / 256, B, 0, stream>>>(zsam, NN * CC, flag);
    k_blend<<<(NN * CC + 255) / 256, B, 0, stream>>>(zsam, logits, flag, zA, NN * CC);

    // 7) K propagation steps, ping-pong zA <-> zB (even step: A->B, odd: B->A)
    for (int s = 0; s < KPROP; ++s) {
        float* zi = (s & 1) ? zB : zA;
        float* zo = (s & 1) ? zA : zB;
        k_prop_init<<<(NN * CC + 255) / 256, B, 0, stream>>>(logits, zo, NN * CC);
        k_prop_scatter<<<(EE * 16 + 255) / 256, B, 0, stream>>>(src, dstv, w, zi, zo);
    }

    // KPROP=10 (even) -> final z in zA
    k_log_softmax<<<(NN + 7) / 8, B, 0, stream>>>(zA, out, NN);
}
